// MambaModel_35459249996632
// MI455X (gfx1250) — matrix operand order repaired and checked by reading, not compiled, not run on hardware
//
#include <hip/hip_runtime.h>
#include <hip/hip_bf16.h>
#include <math.h>
#include <cstdint>

// ---------------- model dims ----------------
#define HDIM  768
#define NLAY  12
#define VOC   50257
#define BATCH 2
#define TLEN  2048
#define NTOK  (BATCH * TLEN)   // 4096
#define EPSLN 1e-5f

typedef __attribute__((ext_vector_type(16))) _Float16 v16h;
typedef __attribute__((ext_vector_type(8)))  _Float16 h8;
typedef __attribute__((ext_vector_type(4)))  _Float16 h4;
typedef __attribute__((ext_vector_type(8)))  float    v8f;

union Frag16 { v16h v; h8 h[2]; };

// ---------------- layernorm: one row per block, writes f16 ----------------
__global__ __launch_bounds__(256)
void ln_kernel(const float* __restrict__ x,
               const float* __restrict__ gamma,
               const float* __restrict__ beta,
               _Float16* __restrict__ out)
{
    __shared__ float red[256];
    const int row = blockIdx.x;
    const int tid = threadIdx.x;
    const float* xr = x + (size_t)row * HDIM;

    float v[3];
    float s = 0.f, s2 = 0.f;
#pragma unroll
    for (int i = 0; i < 3; ++i) {
        float val = xr[tid + i * 256];
        v[i] = val;
        s  += val;
        s2 += val * val;
    }

    red[tid] = s;
    __syncthreads();
    for (int off = 128; off > 0; off >>= 1) {
        if (tid < off) red[tid] += red[tid + off];
        __syncthreads();
    }
    const float mean = red[0] * (1.0f / HDIM);
    __syncthreads();

    red[tid] = s2;
    __syncthreads();
    for (int off = 128; off > 0; off >>= 1) {
        if (tid < off) red[tid] += red[tid + off];
        __syncthreads();
    }
    const float var = red[0] * (1.0f / HDIM) - mean * mean;
    const float inv = rsqrtf(var + EPSLN);

    _Float16* orow = out + (size_t)row * HDIM;
#pragma unroll
    for (int i = 0; i < 3; ++i) {
        int c = tid + i * 256;
        orow[c] = (_Float16)((v[i] - mean) * inv * gamma[c] + beta[c]);
    }
}

// ---------------- f32 -> f16 row converter (optionally strided input) -------
// Handles both compact arrays (ldin == 4*cols4) and strided views (x1 of p).
__global__ __launch_bounds__(256)
void cvt_rows_kernel(const float* __restrict__ in, int ldin,
                     _Float16* __restrict__ out, int ldout,
                     int total4, int cols4)
{
    int i = blockIdx.x * blockDim.x + threadIdx.x;
    if (i >= total4) return;
    int r = i / cols4;
    int c = i - r * cols4;
    float4 v = *(const float4*)(in + (size_t)r * ldin + c * 4);
    h4 h = { (_Float16)v.x, (_Float16)v.y, (_Float16)v.z, (_Float16)v.w };
    *(h4*)(out + (size_t)r * ldout + c * 4) = h;
}

// ---------------- WMMA GEMM: C[M,N] = A[M,K] * B[N,K]^T (+bias) (+=C) ------
// A,B are f16, K-contiguous (lda/ldb in elements). M % 128 == 0, K % 32 == 0,
// N may be ragged (B row index clamped; OOB columns never stored).
// Global -> LDS via GLOBAL_LOAD_ASYNC_TO_LDS_B128, double-buffered.
__global__ __launch_bounds__(256)
void gemm_kernel(const _Float16* __restrict__ A, int lda,
                 const _Float16* __restrict__ B, int ldb,
                 const float* __restrict__ bias,
                 float* __restrict__ C, int ldc,
                 int M, int N, int K, int addC)
{
    __shared__ __align__(16) _Float16 As[2][128][40];   // 32 K-halfs + 8 pad
    __shared__ __align__(16) _Float16 Bs[2][128][40];

    const int tid  = threadIdx.x;
    const int lane = tid & 31;
    const int wid  = tid >> 5;
    const int wm   = wid & 3;     // M sub-block: rows wm*32
    const int wn   = wid >> 2;    // N sub-block: cols wn*64
    const int mBase = blockIdx.y * 128;
    const int nBase = blockIdx.x * 128;

    v8f acc[2][4] = {};

    // A-fragment lane mapping (16-bit A 16x32, ISA 7.12.2)
    const int mfr = lane & 15;
    const int ksA = (lane >> 4) * 8;
    // B-fragment lane mapping (16-bit B 32x16, same k order as A)
    const int kbB = (lane >> 4) * 8;

    const int KT = K >> 5;   // K / 32 tiles

    // Issue one K-tile's async copies: per thread 2 A-chunks + 2 B-chunks of 16B.
    auto issue_tile = [&](int kt, int buf) {
#pragma unroll
        for (int i = 0; i < 2; ++i) {
            int cid = tid + 256 * i;       // 0..511
            int r   = cid >> 2;            // tile row 0..127
            int kq  = cid & 3;             // 8-half group along K
            const _Float16* ga = A + (size_t)(mBase + r) * lda + kt * 32 + kq * 8;
            uint32_t la = (uint32_t)(uintptr_t)&As[buf][r][kq * 8];
            asm volatile("global_load_async_to_lds_b128 %0, %1, off"
                         :: "v"(la), "v"(ga) : "memory");
            int ng = nBase + r;
            if (ng > N - 1) ng = N - 1;    // clamp: no divergence, no OOB reads
            const _Float16* gb = B + (size_t)ng * ldb + kt * 32 + kq * 8;
            uint32_t lb = (uint32_t)(uintptr_t)&Bs[buf][r][kq * 8];
            asm volatile("global_load_async_to_lds_b128 %0, %1, off"
                         :: "v"(lb), "v"(gb) : "memory");
        }
    };

    issue_tile(0, 0);

    for (int kt = 0; kt < KT; ++kt) {
        const int buf = kt & 1;
        if (kt) __syncthreads();                 // all waves done computing on buf^1
        if (kt + 1 < KT) {
            issue_tile(kt + 1, buf ^ 1);         // overlap next tile with compute
            asm volatile("s_wait_asynccnt 0x4" ::: "memory");  // current tile landed
        } else {
            asm volatile("s_wait_asynccnt 0x0" ::: "memory");
        }
        __syncthreads();                         // current tile visible to all waves

        Frag16 af[2];
#pragma unroll
        for (int mi = 0; mi < 2; ++mi) {
            int m = wm * 32 + mi * 16 + mfr;
            af[mi].h[0] = *(const h8*)&As[buf][m][ksA];
            af[mi].h[1] = *(const h8*)&As[buf][m][ksA + 16];
        }
        Frag16 bf[4];
#pragma unroll
        for (int ni = 0; ni < 4; ++ni) {
            int n = wn * 64 + ni * 16 + mfr;
            bf[ni].h[0] = *(const h8*)&Bs[buf][n][kbB];
            bf[ni].h[1] = *(const h8*)&Bs[buf][n][kbB + 16];
        }
#pragma unroll
        for (int mi = 0; mi < 2; ++mi)
#pragma unroll
            for (int ni = 0; ni < 4; ++ni)
                acc[mi][ni] = __builtin_amdgcn_wmma_f32_16x16x32_f16(
                    false, af[mi].v, false, bf[ni].v,
                    (short)0, acc[mi][ni], false, false);
    }

    // epilogue: C/D 16x16 f32 layout -> lane holds rows (lane>>4)*8 .. +7, col lane&15
    const int rBase = (lane >> 4) * 8;
#pragma unroll
    for (int ni = 0; ni < 4; ++ni) {
        int n = nBase + wn * 64 + ni * 16 + (lane & 15);
        if (n >= N) continue;
        float bv = bias ? bias[n] : 0.f;
#pragma unroll
        for (int mi = 0; mi < 2; ++mi) {
            int m0 = mBase + wm * 32 + mi * 16 + rBase;
#pragma unroll
            for (int r = 0; r < 8; ++r) {
                size_t idx = (size_t)(m0 + r) * ldc + n;
                float val = acc[mi][ni][r] + bv;
                if (addC) val += C[idx];
                C[idx] = val;
            }
        }
    }
}

// ---------------- sequential SSM scan + gate fusion -------------------------
__global__ __launch_bounds__(256)
void scan_kernel(const float* __restrict__ p,   // [NTOK, 2H]: x1 | x2
                 const float* __restrict__ Bm,
                 const float* __restrict__ Cm,
                 const float* __restrict__ Av,
                 const float* __restrict__ Dv,
                 float* __restrict__ g)
{
    int gi = blockIdx.x * blockDim.x + threadIdx.x;
    if (gi >= BATCH * HDIM) return;
    const int b = gi / HDIM;
    const int h = gi % HDIM;
    const float a = Av[h];
    const float d = Dv[h];
    float s = 0.f;
    size_t basep = (size_t)b * TLEN * (2 * HDIM) + h;
    size_t baseh = (size_t)b * TLEN * HDIM + h;
    for (int t = 0; t < TLEN; ++t) {
        size_t ip = basep + (size_t)t * (2 * HDIM);
        size_t ih = baseh + (size_t)t * HDIM;
        float xv = p[ip];
        float x2 = p[ip + HDIM];
        float bv = Bm[ih];
        float cv = Cm[ih];
        s = s * a + bv * xv;
        float y = cv * s + d * xv;
        g[ih] = y * (1.f / (1.f + __expf(-x2)));
    }
}

// ---------------- host driver ----------------
static inline void cvt(const float* in, int ldin, _Float16* out, int ldout,
                       int rows, int cols, hipStream_t stream)
{
    int cols4  = cols >> 2;
    int total4 = rows * cols4;
    cvt_rows_kernel<<<(total4 + 255) / 256, 256, 0, stream>>>(
        in, ldin, out, ldout, total4, cols4);
}

extern "C" void kernel_launch(void* const* d_in, const int* in_sizes, int n_in,
                              void* d_out, int out_size, void* d_ws, size_t ws_size,
                              hipStream_t stream)
{
    const float* x_in = (const float*)d_in[0];
    const float* A    = (const float*)d_in[1];
    const float* D    = (const float*)d_in[2];
    const float* Bw   = (const float*)d_in[3];
    const float* Bb   = (const float*)d_in[4];
    const float* Cw   = (const float*)d_in[5];
    const float* Cb   = (const float*)d_in[6];
    const float* ng   = (const float*)d_in[7];
    const float* nb   = (const float*)d_in[8];
    const float* inW  = (const float*)d_in[9];
    const float* inB  = (const float*)d_in[10];
    const float* outW = (const float*)d_in[11];
    const float* outB = (const float*)d_in[12];
    const float* fng  = (const float*)d_in[13];
    const float* fnb  = (const float*)d_in[14];
    const float* lmW  = (const float*)d_in[15];
    const float* lmB  = (const float*)d_in[16];
    float* out = (float*)d_out;

    // workspace carve:
    //   f32: x, p(2H), Bm, Cm, g      f16: act16 (NTOK*H), w16 (VOC*H)
    float* ws = (float*)d_ws;
    float* x  = ws;
    float* p  = x  + (size_t)NTOK * HDIM;
    float* Bm = p  + (size_t)NTOK * 2 * HDIM;
    float* Cm = Bm + (size_t)NTOK * HDIM;
    float* g  = Cm + (size_t)NTOK * HDIM;
    _Float16* act16 = (_Float16*)(g + (size_t)NTOK * HDIM);
    _Float16* w16   = act16 + (size_t)NTOK * HDIM;

    hipMemcpyAsync(x, x_in, (size_t)NTOK * HDIM * sizeof(float),
                   hipMemcpyDeviceToDevice, stream);

    const dim3 blk(256);
    const dim3 gLN(NTOK);
    const dim3 gScan((BATCH * HDIM + 255) / 256);
    const dim3 gIn(2 * HDIM / 128, NTOK / 128);     // 12 x 32
    const dim3 gH(HDIM / 128, NTOK / 128);          //  6 x 32
    const dim3 gLM((VOC + 127) / 128, NTOK / 128);  // 393 x 32

    for (int l = 0; l < NLAY; ++l) {
        // h = LN(x) -> act16 (f16)
        ln_kernel<<<gLN, blk, 0, stream>>>(x, ng + (size_t)l * HDIM,
                                           nb + (size_t)l * HDIM, act16);

        // p = h * inW[l]^T + inB[l]
        cvt(inW + (size_t)l * 2 * HDIM * HDIM, HDIM, w16, HDIM, 2 * HDIM, HDIM, stream);
        gemm_kernel<<<gIn, blk, 0, stream>>>(
            act16, HDIM, w16, HDIM, inB + (size_t)l * 2 * HDIM,
            p, 2 * HDIM, NTOK, 2 * HDIM, HDIM, 0);

        // act16 = f16(x1)  (strided view of p)
        cvt(p, 2 * HDIM, act16, HDIM, NTOK, HDIM, stream);

        // Bm = x1 * Bw[l]^T + Bb[l]
        cvt(Bw + (size_t)l * HDIM * HDIM, HDIM, w16, HDIM, HDIM, HDIM, stream);
        gemm_kernel<<<gH, blk, 0, stream>>>(
            act16, HDIM, w16, HDIM, Bb + (size_t)l * HDIM,
            Bm, HDIM, NTOK, HDIM, HDIM, 0);

        // Cm = x1 * Cw[l]^T + Cb[l]
        cvt(Cw + (size_t)l * HDIM * HDIM, HDIM, w16, HDIM, HDIM, HDIM, stream);
        gemm_kernel<<<gH, blk, 0, stream>>>(
            act16, HDIM, w16, HDIM, Cb + (size_t)l * HDIM,
            Cm, HDIM, NTOK, HDIM, HDIM, 0);

        // scan + gate
        scan_kernel<<<gScan, blk, 0, stream>>>(
            p, Bm, Cm, A + (size_t)l * HDIM, D + (size_t)l * HDIM, g);

        // x += g * outW[l]^T + outB[l]
        cvt(g, HDIM, act16, HDIM, NTOK, HDIM, stream);
        cvt(outW + (size_t)l * HDIM * HDIM, HDIM, w16, HDIM, HDIM, HDIM, stream);
        gemm_kernel<<<gH, blk, 0, stream>>>(
            act16, HDIM, w16, HDIM, outB + (size_t)l * HDIM,
            x, HDIM, NTOK, HDIM, HDIM, 1);
    }

    // final norm + LM head
    ln_kernel<<<gLN, blk, 0, stream>>>(x, fng, fnb, act16);
    cvt(lmW, HDIM, w16, HDIM, VOC, HDIM, stream);
    gemm_kernel<<<gLM, blk, 0, stream>>>(
        act16, HDIM, w16, HDIM, lmB,
        out, VOC, NTOK, VOC, HDIM, 0);
}
